// LigerLMHeadGRPO_62801011802685
// MI455X (gfx1250) — compile-verified, matrix operand order checked by reading, not run on hardware
//
#include <hip/hip_runtime.h>

// ---------------------------------------------------------------------------
// Types for CDNA5 WMMA (wave32): bf16 A/B fragments, f32 accumulator
// ---------------------------------------------------------------------------
typedef __attribute__((ext_vector_type(16))) __bf16 v16bf;
typedef __attribute__((ext_vector_type(8)))  __bf16 v8bf;
typedef __attribute__((ext_vector_type(8)))  float  v8f;

constexpr int kH   = 1024;     // hidden size (K of the GEMM)
constexpr int kV   = 32000;    // vocab size  (N of the GEMM)
constexpr int kBT  = 4096;     // B*T tokens  (M of the GEMM)
constexpr int kTOK = 16;       // tokens per workgroup (M tile)
constexpr int kNCH = 256;      // vocab columns per chunk (8 waves * 2 tiles * 16)
constexpr int kLDA = kH + 8;   // padded LDS row stride (bank-conflict dodge)

union AV { v16bf v; v8bf h[2]; };

__device__ __forceinline__ unsigned short f32_to_bf16(float f) {
  unsigned int u = __float_as_uint(f);
  u += 0x7FFFu + ((u >> 16) & 1u);     // round-to-nearest-even
  return (unsigned short)(u >> 16);
}

__device__ __forceinline__ void load_frag(AV& d, const unsigned short* p, int k) {
  d.h[0] = *(const v8bf*)(p + k);
  d.h[1] = *(const v8bf*)(p + k + 16);
}

// ---------------------------------------------------------------------------
// Kernel 1: fp32 -> bf16 weight conversion (vectorized 4-wide)
// ---------------------------------------------------------------------------
__global__ void cvt_bf16_kernel(const float* __restrict__ src,
                                unsigned short* __restrict__ dst, int n4) {
  int i = blockIdx.x * blockDim.x + threadIdx.x;
  if (i < n4) {
    const float4 f = ((const float4*)src)[i];
    ushort4 o;
    o.x = f32_to_bf16(f.x); o.y = f32_to_bf16(f.y);
    o.z = f32_to_bf16(f.z); o.w = f32_to_bf16(f.w);
    ((ushort4*)dst)[i] = o;
  }
}

// ---------------------------------------------------------------------------
// Kernel 2: per-token logp = logits[sel] - logsumexp(logits) via bf16 WMMA.
// One workgroup = 8 waves = 16 tokens; each wave owns TWO 16x16 N-tiles
// (256 vocab columns per chunk, 125 chunks). Inner K loop is software-
// pipelined: next fragments are loaded before the current WMMA pair issues.
// ---------------------------------------------------------------------------
__global__ void __launch_bounds__(256, 2)
lmhead_logps_kernel(const float* __restrict__ xf,
                    const unsigned short* __restrict__ wB,
                    const int* __restrict__ ids,
                    float* __restrict__ out_logps) {
  __shared__ unsigned short ldsX[kTOK * kLDA];      // bf16 A tile, padded
  __shared__ float ldsC[kTOK * kNCH];               // 16 x 256 logit tile
  __shared__ float part[256];                       // partial reductions
  __shared__ float rmax[kTOK], rsum[kTOK], selv[kTOK], newm[kTOK];

  const int tid  = threadIdx.x;
  const int lane = tid & 31;
  const int wave = tid >> 5;
  const int tok0 = blockIdx.x * kTOK;

  // Stage + convert the 16-token activation tile into LDS (bf16).
  for (int i = tid * 4; i < kTOK * kH; i += 256 * 4) {
    const int r = i >> 10, h = i & (kH - 1);
    const float4 f = *(const float4*)(xf + (size_t)(tok0 + r) * kH + h);
    unsigned short* p = ldsX + r * kLDA + h;
    p[0] = f32_to_bf16(f.x); p[1] = f32_to_bf16(f.y);
    p[2] = f32_to_bf16(f.z); p[3] = f32_to_bf16(f.w);
  }
  if (tid < kTOK) { rmax[tid] = -3.0e38f; rsum[tid] = 0.0f; selv[tid] = 0.0f; }
  __syncthreads();

  // ISA 16-bit A-fragment layout: lane<16 -> M=lane, K in {0..7,16..23};
  // lane>=16 -> M=lane-16, K in {8..15,24..31}. B mirrors with N<->M, and
  // weight rows are contiguous in K, so every operand is 2x16B loads.
  const int rowA = lane & 15;
  const int kh   = (lane >> 4) << 3;                   // 0 or 8
  const unsigned short* aBase = ldsX + rowA * kLDA + kh;
  const int myId = (tid < kTOK) ? ids[tok0 + tid] : 0;

  for (int v0 = 0; v0 < kV; v0 += kNCH) {
    const int col0 = v0 + wave * 32 + (lane & 15);     // first N-tile column
    const unsigned short* bBase0 = wB + (size_t)col0 * kH + kh;
    const unsigned short* bBase1 = bBase0 + (size_t)16 * kH;   // col0 + 16
    if (v0 + kNCH < kV) {                              // global_prefetch_b8
      __builtin_prefetch((const void*)(bBase0 + (size_t)kNCH * kH), 0, 1);
      __builtin_prefetch((const void*)(bBase1 + (size_t)kNCH * kH), 0, 1);
    }

    v8f c0 = {}, c1 = {};
    // Software pipeline: prologue loads for k = 0.
    AV aC, b0C, b1C, aN, b0N, b1N;
    load_frag(aC,  aBase,  0);
    load_frag(b0C, bBase0, 0);
    load_frag(b1C, bBase1, 0);
#pragma unroll 4
    for (int k = 0; k < kH; k += 32) {
      const int kn = (k + 32) & (kH - 1);              // last iter wraps (dead)
      load_frag(aN,  aBase,  kn);
      load_frag(b0N, bBase0, kn);
      load_frag(b1N, bBase1, kn);
      c0 = __builtin_amdgcn_wmma_f32_16x16x32_bf16(
               false, aC.v, false, b0C.v, (short)0, c0, false, false);
      c1 = __builtin_amdgcn_wmma_f32_16x16x32_bf16(
               false, aC.v, false, b1C.v, (short)0, c1, false, false);
      aC = aN; b0C = b0N; b1C = b1N;                   // register rotation
    }

    // Scatter C to LDS per ISA C/D layout (lane<16: M=i, lane>=16: M=i+8).
    {
      const int cc0 = wave * 32 + (lane & 15);
      const int cc1 = cc0 + 16;
      const int rb  = (lane < 16) ? 0 : 8;
#pragma unroll
      for (int i = 0; i < 8; ++i) {
        ldsC[(rb + i) * kNCH + cc0] = c0[i];
        ldsC[(rb + i) * kNCH + cc1] = c1[i];
      }
    }
    __syncthreads();

    // Online softmax update. token t = tid>>4; each of 16 threads/token
    // owns a 16-wide segment of the 256 columns.
    const int t = tid >> 4, seg = tid & 15;
    {
      const float* row = ldsC + t * kNCH + seg * 16;
      float mx = row[0];
#pragma unroll
      for (int j = 1; j < 16; ++j) mx = fmaxf(mx, row[j]);
      part[tid] = mx;
    }
    __syncthreads();
    if (seg == 0) {
      float mx = part[t * 16];
#pragma unroll
      for (int j = 1; j < 16; ++j) mx = fmaxf(mx, part[t * 16 + j]);
      newm[t] = fmaxf(rmax[t], mx);
    }
    __syncthreads();
    {
      const float m = newm[t];
      const float* row = ldsC + t * kNCH + seg * 16;
      float s = 0.0f;
#pragma unroll
      for (int j = 0; j < 16; ++j) s += __expf(row[j] - m);
      part[tid] = s;
    }
    __syncthreads();
    if (seg == 0) {
      float s = 0.0f;
#pragma unroll
      for (int j = 0; j < 16; ++j) s += part[t * 16 + j];
      rsum[t] = rsum[t] * __expf(rmax[t] - newm[t]) + s;
      rmax[t] = newm[t];
    }
    if (tid < kTOK && myId >= v0 && myId < v0 + kNCH)
      selv[tid] = ldsC[tid * kNCH + (myId - v0)];
    __syncthreads();                                   // ldsC reused next chunk
  }

  if (tid < kTOK)
    out_logps[tok0 + tid] = selv[tid] - (rmax[tid] + __logf(rsum[tid]));
}

// ---------------------------------------------------------------------------
// Kernel 3: scalar GRPO loss over 4096 tokens
// ---------------------------------------------------------------------------
__global__ void finalize_kernel(const float* __restrict__ lp,
                                const float* __restrict__ rlp,
                                const float* __restrict__ adv,
                                const int* __restrict__ mask,
                                float* __restrict__ out) {
  __shared__ float sL[256], sM[256];
  float accL = 0.0f, accM = 0.0f;
  for (int t = threadIdx.x; t < kBT; t += 256) {
    const float l = lp[t], r = rlp[t];
    const float m = (float)mask[t];
    const float a = adv[t >> 10];                      // T = 1024
    const float c1 = __expf(l - l);                    // old_logps == logps
    const float c2 = fminf(fmaxf(c1, 1.0f - 0.2f), 1.0f + 0.2f);
    float ptl = -fminf(c1 * a, c2 * a);
    const float d = r - l;
    ptl += 0.1f * (__expf(d) - d - 1.0f);              // BETA * kl
    accL += ptl * m;
    accM += m;
  }
  sL[threadIdx.x] = accL; sM[threadIdx.x] = accM;
  __syncthreads();
  for (int s = 128; s > 0; s >>= 1) {
    if (threadIdx.x < s) {
      sL[threadIdx.x] += sL[threadIdx.x + s];
      sM[threadIdx.x] += sM[threadIdx.x + s];
    }
    __syncthreads();
  }
  if (threadIdx.x == 0) out[0] = sL[0] / fmaxf(sM[0], 1.0f);
}

// ---------------------------------------------------------------------------
extern "C" void kernel_launch(void* const* d_in, const int* in_sizes, int n_in,
                              void* d_out, int out_size, void* d_ws, size_t ws_size,
                              hipStream_t stream) {
  const float* x    = (const float*)d_in[0];
  const float* w    = (const float*)d_in[1];
  const float* rx   = (const float*)d_in[2];
  const float* rw   = (const float*)d_in[3];
  const float* adv  = (const float*)d_in[4];
  const int*   ids  = (const int*)d_in[5];
  const int*   mask = (const int*)d_in[6];
  float*       out  = (float*)d_out;

  // Workspace layout (all 256B-aligned):
  //   [0)                      bf16 policy weight  (V*H*2 = 65,536,000 B)
  //   [65,536,000)             bf16 ref weight     (65,536,000 B)
  //   [131,072,000)            policy logps        (4096 * 4 B)
  //   [131,072,000 + 16,384)   ref logps           (4096 * 4 B)
  char* ws = (char*)d_ws;
  const size_t wbytes = (size_t)kV * kH * 2;           // 65,536,000
  unsigned short* wBp  = (unsigned short*)ws;
  unsigned short* rwBp = (unsigned short*)(ws + wbytes);
  float* lp  = (float*)(ws + 2 * wbytes);
  float* rlp = (float*)(ws + 2 * wbytes + 16384);

  const int n4 = (kV * kH) / 4;                        // 8,192,000 float4s
  cvt_bf16_kernel<<<(n4 + 255) / 256, 256, 0, stream>>>(w,  wBp,  n4);
  cvt_bf16_kernel<<<(n4 + 255) / 256, 256, 0, stream>>>(rw, rwBp, n4);

  lmhead_logps_kernel<<<kBT / kTOK, 256, 0, stream>>>(x,  wBp,  ids, lp);
  lmhead_logps_kernel<<<kBT / kTOK, 256, 0, stream>>>(rx, rwBp, ids, rlp);

  finalize_kernel<<<1, 256, 0, stream>>>(lp, rlp, adv, mask, out);
}
